// Extractor_85280870630092
// MI455X (gfx1250) — compile-verified, matrix operand order checked by reading, not run on hardware
//
#include <hip/hip_runtime.h>
#include <hip/hip_bf16.h>
#include <cstdint>
#include <cstddef>

// ---------------------------------------------------------------- constants
static constexpr int D  = 768;
static constexpr int DM = 3072;
static constexpr int H  = 12;
static constexpr int B  = 32;
static constexpr int TX = 16;
static constexpr int TY = 4096;

typedef __attribute__((ext_vector_type(16))) __bf16        bf16x16;
typedef __attribute__((ext_vector_type(8)))  float         f32x8;
typedef __attribute__((ext_vector_type(8)))  unsigned int  u32x8;
typedef int b128_t __attribute__((vector_size(16)));        // matches async-LDS builtin pointee

#if __has_builtin(__builtin_amdgcn_global_load_async_to_lds_b128) && \
    __has_builtin(__builtin_amdgcn_s_wait_asynccnt)
#define HAVE_ASYNC_LDS 1
#endif

#define DEV static __device__ __forceinline__

DEV unsigned short f2bf(float f) {
    return __builtin_bit_cast(unsigned short, (__bf16)f);   // native v_cvt, RTNE
}

DEV f32x8 wmma_bf16(bf16x16 a, bf16x16 b, f32x8 c) {
    // D = A(16x32 bf16) * B(32x16 bf16) + C(16x16 f32)
    return __builtin_amdgcn_wmma_f32_16x16x32_bf16(false, a, false, b,
                                                   (short)0, c, false, false);
}

// A/B fragment loader for 16x16x32 bf16 WMMA (ISA 7.12.2 layout).
// lane&15 selects the row (A: M-row, B: N-column of W where B[k][n]=W[n][k]).
// elements 0..7  = k of 8*(lane>>4) .. +7 ; elements 8..15 = 16 + same.
DEV bf16x16 load_frag(const unsigned short* base, int ld, int lane) {
    const unsigned short* p = base + (size_t)(lane & 15) * ld + ((lane >> 4) & 1) * 8;
    uint4 a  = *(const uint4*)(p);
    uint4 bq = *(const uint4*)(p + 16);
    u32x8 r;
    r[0] = a.x;  r[1] = a.y;  r[2] = a.z;  r[3] = a.w;
    r[4] = bq.x; r[5] = bq.y; r[6] = bq.z; r[7] = bq.w;
    return __builtin_bit_cast(bf16x16, r);
}

#define ZERO8 {0.f,0.f,0.f,0.f,0.f,0.f,0.f,0.f}

// ---------------------------------------------------------------- fp32 -> bf16
__global__ __launch_bounds__(256) void f2bf_kernel(const float* __restrict__ in,
                                                   unsigned short* __restrict__ out,
                                                   int n) {
    int i = (blockIdx.x * 256 + threadIdx.x) * 4;
#pragma unroll
    for (int j = 0; j < 4; ++j)
        if (i + j < n) out[i + j] = f2bf(in[i + j]);
}

// ---------------------------------------------------------------- LN -> bf16
__global__ __launch_bounds__(256) void ln_kernel(const float* __restrict__ in,
                                                 const float* __restrict__ w,
                                                 const float* __restrict__ bparm,
                                                 unsigned short* __restrict__ out) {
    int lane = threadIdx.x & 31, wv = threadIdx.x >> 5;
#pragma unroll 1
    for (int i = 0; i < 8; ++i) {
        size_t row = (size_t)blockIdx.x * 64 + wv * 8 + i;
        const float* r = in + row * D;
        float v[24], s = 0.f, s2 = 0.f;
#pragma unroll
        for (int t = 0; t < 24; ++t) { float x = r[lane + 32 * t]; v[t] = x; s += x; s2 += x * x; }
#pragma unroll
        for (int m = 1; m < 32; m <<= 1) { s += __shfl_xor(s, m, 32); s2 += __shfl_xor(s2, m, 32); }
        float mean = s * (1.f / 768.f);
        float rstd = rsqrtf(s2 * (1.f / 768.f) - mean * mean + 1e-5f);
#pragma unroll
        for (int t = 0; t < 24; ++t) {
            int k = lane + 32 * t;
            out[row * D + k] = f2bf((v[t] - mean) * rstd * w[k] + bparm[k]);
        }
    }
}

// ---------------------------------------------------------------- LN3 + K/V projection
// grid: 32 (b) * 128 (ty tiles of 32).  K/V stored bf16 [b][h][ty][64].
// Each wave: 2x2 register tile (2 A-tiles x 2 W-tiles -> 4 WMMAs per 4 frag loads).
__global__ __launch_bounds__(256) void kvproj_kernel(const float* __restrict__ y,
                                                     const float* __restrict__ lnw,
                                                     const float* __restrict__ lnb,
                                                     const unsigned short* __restrict__ wkv,
                                                     const float* __restrict__ inb,
                                                     unsigned short* __restrict__ Kbuf,
                                                     unsigned short* __restrict__ Vbuf) {
    __shared__ __align__(16) unsigned short xs[32 * 776];
    int lane = threadIdx.x & 31, wv = threadIdx.x >> 5;
    int b   = blockIdx.x >> 7;
    int ty0 = (blockIdx.x & 127) * 32;

    // fused LayerNorm(ln3) of 32 y-rows into LDS (bf16)
#pragma unroll 1
    for (int i = 0; i < 4; ++i) {
        int m = wv * 4 + i;
        const float* yr = y + ((size_t)(ty0 + m) * B + b) * D;
        float v[24], s = 0.f, s2 = 0.f;
#pragma unroll
        for (int t = 0; t < 24; ++t) { float x = yr[lane + 32 * t]; v[t] = x; s += x; s2 += x * x; }
#pragma unroll
        for (int msk = 1; msk < 32; msk <<= 1) { s += __shfl_xor(s, msk, 32); s2 += __shfl_xor(s2, msk, 32); }
        float mean = s * (1.f / 768.f);
        float rstd = rsqrtf(s2 * (1.f / 768.f) - mean * mean + 1e-5f);
#pragma unroll
        for (int t = 0; t < 24; ++t) {
            int k = lane + 32 * t;
            xs[m * 776 + k] = f2bf((v[t] - mean) * rstd * lnw[k] + lnb[k]);
        }
    }
    __syncthreads();

    int qc = lane & 15, hh = lane >> 4;
#pragma unroll 1
    for (int np = 0; np < 6; ++np) {
        int n0 = wv * 192 + np * 32;                // n in [0,1536): K rows then V rows
        const unsigned short* Wr0 = wkv + (size_t)n0 * D;
        const unsigned short* Wr1 = wkv + (size_t)(n0 + 16) * D;
        f32x8 c[2][2] = {{ZERO8, ZERO8}, {ZERO8, ZERO8}};   // [m-tile][n-tile]
#pragma unroll 1
        for (int kk = 0; kk < D; kk += 32) {
            bf16x16 b0 = load_frag(Wr0 + kk, D, lane);
            bf16x16 b1 = load_frag(Wr1 + kk, D, lane);
            bf16x16 a0 = load_frag(xs + kk, 776, lane);
            bf16x16 a1 = load_frag(xs + 16 * 776 + kk, 776, lane);
            c[0][0] = wmma_bf16(a0, b0, c[0][0]);
            c[0][1] = wmma_bf16(a0, b1, c[0][1]);
            c[1][0] = wmma_bf16(a1, b0, c[1][0]);
            c[1][1] = wmma_bf16(a1, b1, c[1][1]);
        }
#pragma unroll
        for (int ni = 0; ni < 2; ++ni) {
            int n = n0 + ni * 16 + qc;
            float bias = inb[D + n];                // covers K then V bias ranges
            unsigned short* dst = (n < D) ? Kbuf : Vbuf;
            int h2 = ((n < D) ? n : (n - D)) >> 6;
            int hd = n & 63;
            size_t basead = (size_t)(b * H + h2) * TY * 64 + hd;
#pragma unroll
            for (int r = 0; r < 8; ++r) {
                int m = hh * 8 + r;
                dst[basead + (size_t)(ty0 + m) * 64]      = f2bf(c[0][ni][r] + bias);
                dst[basead + (size_t)(ty0 + 16 + m) * 64] = f2bf(c[1][ni][r] + bias);
            }
        }
    }
}

// ---------------------------------------------------------------- LN1 + Q projection (scaled by 1/8)
__global__ __launch_bounds__(256) void qproj_kernel(const float* __restrict__ x,
                                                    const float* __restrict__ lnw,
                                                    const float* __restrict__ lnb,
                                                    const unsigned short* __restrict__ wq,
                                                    const float* __restrict__ inb,
                                                    unsigned short* __restrict__ qbuf) {
    __shared__ __align__(16) unsigned short xs[16 * 776];
    int lane = threadIdx.x & 31, wv = threadIdx.x >> 5;
    int b = blockIdx.x;
#pragma unroll 1
    for (int i = 0; i < 2; ++i) {
        int m = wv * 2 + i;
        const float* xr = x + ((size_t)m * B + b) * D;
        float v[24], s = 0.f, s2 = 0.f;
#pragma unroll
        for (int t = 0; t < 24; ++t) { float q = xr[lane + 32 * t]; v[t] = q; s += q; s2 += q * q; }
#pragma unroll
        for (int msk = 1; msk < 32; msk <<= 1) { s += __shfl_xor(s, msk, 32); s2 += __shfl_xor(s2, msk, 32); }
        float mean = s * (1.f / 768.f);
        float rstd = rsqrtf(s2 * (1.f / 768.f) - mean * mean + 1e-5f);
#pragma unroll
        for (int t = 0; t < 24; ++t) {
            int k = lane + 32 * t;
            xs[m * 776 + k] = f2bf((v[t] - mean) * rstd * lnw[k] + lnb[k]);
        }
    }
    __syncthreads();

    int qc = lane & 15, hh = lane >> 4;
#pragma unroll 1
    for (int nt = 0; nt < 6; ++nt) {
        int n0 = wv * 96 + nt * 16;
        const unsigned short* Wr = wq + (size_t)n0 * D;
        f32x8 c = ZERO8;
#pragma unroll 1
        for (int kk = 0; kk < D; kk += 32)
            c = wmma_bf16(load_frag(xs + kk, 776, lane), load_frag(Wr + kk, D, lane), c);
        int n = n0 + qc;
        float bias = inb[n];
        int h2 = n >> 6, hd = n & 63;
#pragma unroll
        for (int r = 0; r < 8; ++r) {
            int m = hh * 8 + r;
            qbuf[((size_t)(b * H + h2) * TX + m) * 64 + hd] = f2bf(0.125f * (c[r] + bias));
        }
    }
}

// ---------------------------------------------------------------- flash attention, one block per (b,h)
// Transposed-scores scheme: S^T = K*q^T, out^T = V^T * P^T; the S^T C-tiles are
// bit-compatible with the B-fragment layout of the second WMMA (no LDS transpose of P).
// V chunks staged memory->LDS with GLOBAL_LOAD_ASYNC_TO_LDS (ASYNCcnt) when available.
__global__ __launch_bounds__(256) void attn_kernel(const unsigned short* __restrict__ qbuf,
                                                   const unsigned short* __restrict__ Kbuf,
                                                   const unsigned short* __restrict__ Vbuf,
                                                   unsigned short* __restrict__ attn) {
    __shared__ __align__(16) float smemf[9216];          // 36864 B, reused
    int lane = threadIdx.x & 31, wv = threadIdx.x >> 5;
    int bh = blockIdx.x, b = bh / H, h = bh % H;
    int qc = lane & 15, hh = lane >> 4;

    const unsigned short* qp = qbuf + (size_t)bh * TX * 64;
    const unsigned short* Kp = Kbuf + (size_t)bh * TY * 64;
    const unsigned short* Vp = Vbuf + (size_t)bh * TY * 64;

    bf16x16 q0 = load_frag(qp + 0, 64, lane);            // B-frags of q^T, hd slices 0..31 / 32..63
    bf16x16 q1 = load_frag(qp + 32, 64, lane);

    f32x8 acc[4] = {ZERO8, ZERO8, ZERO8, ZERO8};         // out^T tiles [hd-tile][.]
    float mrun = -3.4e38f, lrun = 0.f;
    unsigned short* vb = (unsigned short*)smemf + (size_t)wv * (32 * 72);  // per-wave V stage

#pragma unroll 1
    for (int c = wv; c < 128; c += 8) {                  // 32-key chunks, split-K over 8 waves
        int key0 = c * 32;
        // stage V chunk [32 keys][64] into LDS (lane = key, padded stride 72)
#ifdef HAVE_ASYNC_LDS
        {
            const unsigned short* gsrc = Vp + (size_t)(key0 + lane) * 64;
            unsigned short* ldst = vb + lane * 72;
#pragma unroll
            for (int j = 0; j < 8; ++j)
                __builtin_amdgcn_global_load_async_to_lds_b128(
                    (__attribute__((address_space(1))) b128_t*)(gsrc + j * 8),
                    (__attribute__((address_space(3))) b128_t*)(ldst + j * 8), 0, 0);
            __builtin_amdgcn_s_wait_asynccnt(0);
        }
#else
        {
            const uint4* src = (const uint4*)(Vp + (size_t)(key0 + lane) * 64);
            uint4* dstp = (uint4*)(vb + lane * 72);
#pragma unroll
            for (int j = 0; j < 8; ++j) dstp[j] = src[j];
        }
#endif
        __builtin_prefetch(Kp + (size_t)(key0 + 256) * 64, 0, 0);
        __builtin_prefetch(Vp + (size_t)(key0 + 256) * 64, 0, 0);

        // S^T tiles: 2 key-tiles x (2 hd-slices)
        f32x8 s0 = ZERO8, s1 = ZERO8;
        s0 = wmma_bf16(load_frag(Kp + (size_t)key0 * 64, 64, lane), q0, s0);
        s0 = wmma_bf16(load_frag(Kp + (size_t)key0 * 64 + 32, 64, lane), q1, s0);
        s1 = wmma_bf16(load_frag(Kp + (size_t)(key0 + 16) * 64, 64, lane), q0, s1);
        s1 = wmma_bf16(load_frag(Kp + (size_t)(key0 + 16) * 64 + 32, 64, lane), q1, s1);

        // online softmax per q column (= per lane&15)
        float lm = s0[0];
#pragma unroll
        for (int r = 1; r < 8; ++r) lm = fmaxf(lm, s0[r]);
#pragma unroll
        for (int r = 0; r < 8; ++r) lm = fmaxf(lm, s1[r]);
        lm = fmaxf(lm, __shfl_xor(lm, 16, 32));
        float mnew = fmaxf(mrun, lm);
        float corr = __expf(mrun - mnew);
        float p0[8], p1[8], rs = 0.f;
#pragma unroll
        for (int r = 0; r < 8; ++r) { p0[r] = __expf(s0[r] - mnew); rs += p0[r]; }
#pragma unroll
        for (int r = 0; r < 8; ++r) { p1[r] = __expf(s1[r] - mnew); rs += p1[r]; }
        rs += __shfl_xor(rs, 16, 32);
        lrun = lrun * corr + rs;
        mrun = mnew;
#pragma unroll
        for (int mt = 0; mt < 4; ++mt)
#pragma unroll
            for (int r = 0; r < 8; ++r) acc[mt][r] *= corr;

        // B-frag of P^T straight from the score registers
        u32x8 pr;
#pragma unroll
        for (int i = 0; i < 4; ++i) {
            pr[i]     = (unsigned)f2bf(p0[2 * i]) | ((unsigned)f2bf(p0[2 * i + 1]) << 16);
            pr[4 + i] = (unsigned)f2bf(p1[2 * i]) | ((unsigned)f2bf(p1[2 * i + 1]) << 16);
        }
        bf16x16 bP = __builtin_bit_cast(bf16x16, pr);

        int kb = hh * 8;
#pragma unroll
        for (int mt = 0; mt < 4; ++mt) {                 // A = V^T slices via LDS gather
            int hd = mt * 16 + qc;
            u32x8 vr;
#pragma unroll
            for (int i = 0; i < 4; ++i) {
                vr[i]     = (unsigned)vb[(kb + 2 * i) * 72 + hd]      | ((unsigned)vb[(kb + 2 * i + 1) * 72 + hd] << 16);
                vr[4 + i] = (unsigned)vb[(16 + kb + 2 * i) * 72 + hd] | ((unsigned)vb[(16 + kb + 2 * i + 1) * 72 + hd] << 16);
            }
            acc[mt] = wmma_bf16(__builtin_bit_cast(bf16x16, vr), bP, acc[mt]);
        }
    }

    __syncthreads();                                     // everyone done with vb
    float* accS = smemf;                                 // [8][16 q][64 hd]
    float* mS   = smemf + 8192;
    float* lS   = mS + 128;
    float* accw = accS + wv * 1024;
#pragma unroll
    for (int mt = 0; mt < 4; ++mt)
#pragma unroll
        for (int r = 0; r < 8; ++r)
            accw[qc * 64 + mt * 16 + hh * 8 + r] = acc[mt][r];
    if (lane < 16) { mS[wv * 16 + lane] = mrun; lS[wv * 16 + lane] = lrun; }
    __syncthreads();

    // combine 8 split-K partials; 4 output elements per thread
#pragma unroll 1
    for (int e = threadIdx.x * 4; e < threadIdx.x * 4 + 4; ++e) {
        int q = e >> 6, hd = e & 63;
        float M = mS[q];
#pragma unroll
        for (int w8 = 1; w8 < 8; ++w8) M = fmaxf(M, mS[w8 * 16 + q]);
        float sum = 0.f, val = 0.f;
#pragma unroll
        for (int w8 = 0; w8 < 8; ++w8) {
            float f = __expf(mS[w8 * 16 + q] - M);
            sum += lS[w8 * 16 + q] * f;
            val += accS[w8 * 1024 + q * 64 + hd] * f;
        }
        attn[((size_t)q * B + b) * D + h * 64 + hd] = f2bf(val / sum);
    }
}

// ---------------------------------------------------------------- small WMMA GEMM (out_proj / fc / proj)
// C[m][n] = act( sum_k A[m][k]*W[n][k] + bias[n] ) (+ resid[m][n]); 128x64 block tile,
// each wave owns a 32x32 register tile (4 WMMAs per 4 fragment loads).
template <int ACT, bool RESID, bool OUTBF>
__global__ __launch_bounds__(256) void gemm_kernel(const unsigned short* __restrict__ A, int K,
                                                   const unsigned short* __restrict__ W,
                                                   const float* __restrict__ bias,
                                                   const float* __restrict__ resid,
                                                   void* __restrict__ outp, int N) {
    int lane = threadIdx.x & 31, wv = threadIdx.x >> 5;
    int n0 = blockIdx.x * 64 + (wv >> 2) * 32;
    int m0 = blockIdx.y * 128 + (wv & 3) * 32;
    const unsigned short* Ar0 = A + (size_t)m0 * K;
    const unsigned short* Ar1 = A + (size_t)(m0 + 16) * K;
    const unsigned short* Wr0 = W + (size_t)n0 * K;
    const unsigned short* Wr1 = W + (size_t)(n0 + 16) * K;
    f32x8 c[2][2] = {{ZERO8, ZERO8}, {ZERO8, ZERO8}};       // [m-tile][n-tile]
#pragma unroll 1
    for (int kk = 0; kk < K; kk += 32) {
        bf16x16 a0 = load_frag(Ar0 + kk, K, lane);
        bf16x16 a1 = load_frag(Ar1 + kk, K, lane);
        bf16x16 b0 = load_frag(Wr0 + kk, K, lane);
        bf16x16 b1 = load_frag(Wr1 + kk, K, lane);
        c[0][0] = wmma_bf16(a0, b0, c[0][0]);
        c[0][1] = wmma_bf16(a0, b1, c[0][1]);
        c[1][0] = wmma_bf16(a1, b0, c[1][0]);
        c[1][1] = wmma_bf16(a1, b1, c[1][1]);
    }
    int qc = lane & 15, hh = lane >> 4;
#pragma unroll
    for (int mi = 0; mi < 2; ++mi)
#pragma unroll
        for (int r = 0; r < 8; ++r) {
            int m = m0 + mi * 16 + hh * 8 + r;
#pragma unroll
            for (int ni = 0; ni < 2; ++ni) {
                int n = n0 + ni * 16 + qc;
                float v = c[mi][ni][r] + bias[n];
                if (ACT == 1) v = v / (1.f + __expf(-1.702f * v));   // quickGELU
                if (RESID) v += resid[(size_t)m * N + n];
                if (OUTBF) ((unsigned short*)outp)[(size_t)m * N + n] = f2bf(v);
                else       ((float*)outp)[(size_t)m * N + n] = v;
            }
        }
}

// ---------------------------------------------------------------- host launcher
extern "C" void kernel_launch(void* const* d_in, const int* in_sizes, int n_in,
                              void* d_out, int out_size, void* d_ws, size_t ws_size,
                              hipStream_t stream) {
    const float* x     = (const float*)d_in[0];
    const float* y     = (const float*)d_in[1];
    const float* inw   = (const float*)d_in[2];
    const float* inb   = (const float*)d_in[3];
    const float* outw  = (const float*)d_in[4];
    const float* outb  = (const float*)d_in[5];
    const float* ln1w  = (const float*)d_in[6];
    const float* ln1b  = (const float*)d_in[7];
    const float* ln2w  = (const float*)d_in[8];
    const float* ln2b  = (const float*)d_in[9];
    const float* ln3w  = (const float*)d_in[10];
    const float* ln3b  = (const float*)d_in[11];
    const float* fcw   = (const float*)d_in[12];
    const float* fcb   = (const float*)d_in[13];
    const float* projw = (const float*)d_in[14];
    const float* projb = (const float*)d_in[15];
    float* out = (float*)d_out;
    (void)in_sizes; (void)n_in; (void)out_size; (void)ws_size;

    char* ws = (char*)d_ws;
    size_t off = 0;
    auto alloc = [&](size_t bytes) -> void* {
        void* p = ws + off;
        off += (bytes + 255) & ~(size_t)255;
        return p;
    };
    const size_t NKV = (size_t)B * H * TY * 64;              // 100,663,296 elems
    unsigned short* win_bf   = (unsigned short*)alloc((size_t)3 * D * D * 2);
    unsigned short* wout_bf  = (unsigned short*)alloc((size_t)D * D * 2);
    unsigned short* wfc_bf   = (unsigned short*)alloc((size_t)DM * D * 2);
    unsigned short* wproj_bf = (unsigned short*)alloc((size_t)D * DM * 2);
    unsigned short* Kbuf     = (unsigned short*)alloc(NKV * 2);
    unsigned short* Vbuf     = (unsigned short*)alloc(NKV * 2);
    unsigned short* qbuf     = (unsigned short*)alloc((size_t)B * H * TX * 64 * 2);
    unsigned short* attn_bf  = (unsigned short*)alloc((size_t)512 * D * 2);
    float*          x2       = (float*)alloc((size_t)512 * D * 4);
    unsigned short* x2n_bf   = (unsigned short*)alloc((size_t)512 * D * 2);
    unsigned short* g_bf     = (unsigned short*)alloc((size_t)512 * DM * 2);

    auto blocks = [](int n) { return (n + 1023) / 1024; };
    f2bf_kernel<<<blocks(3 * D * D), 256, 0, stream>>>(inw,   win_bf,   3 * D * D);
    f2bf_kernel<<<blocks(D * D),     256, 0, stream>>>(outw,  wout_bf,  D * D);
    f2bf_kernel<<<blocks(DM * D),    256, 0, stream>>>(fcw,   wfc_bf,   DM * D);
    f2bf_kernel<<<blocks(D * DM),    256, 0, stream>>>(projw, wproj_bf, D * DM);

    kvproj_kernel<<<B * (TY / 32), 256, 0, stream>>>(y, ln3w, ln3b,
                                                     win_bf + (size_t)D * D, inb, Kbuf, Vbuf);
    qproj_kernel<<<B, 256, 0, stream>>>(x, ln1w, ln1b, win_bf, inb, qbuf);
    attn_kernel<<<B * H, 256, 0, stream>>>(qbuf, Kbuf, Vbuf, attn_bf);

    // x2 = x + attn @ out_w^T + out_b
    gemm_kernel<0, true, false><<<dim3(D / 64, 512 / 128), 256, 0, stream>>>(
        attn_bf, D, wout_bf, outb, x, x2, D);
    // ln2(x2) -> bf16
    ln_kernel<<<512 / 64, 256, 0, stream>>>(x2, ln2w, ln2b, x2n_bf);
    // g = quickgelu(x2n @ fc_w^T + fc_b)
    gemm_kernel<1, false, true><<<dim3(DM / 64, 512 / 128), 256, 0, stream>>>(
        x2n_bf, D, wfc_bf, fcb, nullptr, g_bf, DM);
    // out = x2 + g @ proj_w^T + proj_b
    gemm_kernel<0, true, false><<<dim3(D / 64, 512 / 128), 256, 0, stream>>>(
        g_bf, DM, wproj_bf, projb, x2, out, D);
}
